// DynamicNeuralTuringMachine_77498389889286
// MI455X (gfx1250) — compile-verified, matrix operand order checked by reading, not run on hardware
//
#include <hip/hip_runtime.h>
#include <math.h>

typedef __bf16 bf16_t;
typedef __attribute__((ext_vector_type(16))) __bf16 v16bf;
typedef __attribute__((ext_vector_type(8)))  __bf16 v8bf;
typedef __attribute__((ext_vector_type(8)))  float  v8f;

#define B_ 64
#define S_ 32
#define F_ 512
#define N_ 512
#define C_ 224
#define A_ 32
#define M_ 256
#define H_ 1024
#define O_ 256
#define EPS_ 1e-8f

// ---------------------------------------------------------------------------
// WMMA bf16 GEMM:  C[m,n] (+)= sum_k A[m,k] * W[n,k]   (i.e. C = A @ W^T)
// A: (16*mtiles) x K row-major bf16, W: (16*ntiles) x K row-major bf16.
// One 16x16 output tile per wave; K looped in chunks of 32.
// ---------------------------------------------------------------------------
__device__ __forceinline__ v16bf load_frag16(const bf16_t* tile, int ld, int lane) {
  // A-matrix 16x32 bf16 layout (ISA 7.12.2): lane&15 = row; lanes 0-15 hold
  // K = 0..7 (elems 0..7) and 16..23 (elems 8..15); lanes 16-31 hold K=8..15, 24..31.
  int r  = lane & 15;
  int kh = (lane >> 4) << 3;
  const bf16_t* p = tile + (size_t)r * ld + kh;
  v8bf lo = *reinterpret_cast<const v8bf*>(p);
  v8bf hi = *reinterpret_cast<const v8bf*>(p + 16);
  v16bf o;
#pragma unroll
  for (int i = 0; i < 8; ++i) { o[i] = lo[i]; o[i + 8] = hi[i]; }
  return o;
}

__global__ void wmma_gemm_bf16(const bf16_t* __restrict__ A, int lda,
                               const bf16_t* __restrict__ W, int ldw,
                               float* __restrict__ C, int ldc,
                               const float* __restrict__ bias, int accum,
                               int mtiles, int ntiles, int K) {
  int wave = (int)((blockIdx.x * blockDim.x + threadIdx.x) >> 5);
  int lane = (int)(threadIdx.x & 31);
  if (wave >= mtiles * ntiles) return;          // whole-wave exit: EXEC stays all-ones
  int tm = wave % mtiles, tn = wave / mtiles;
  const bf16_t* Abase = A + (size_t)(tm * 16) * lda;
  const bf16_t* Wbase = W + (size_t)(tn * 16) * ldw;
  int col = tn * 16 + (lane & 15);
  int rb  = tm * 16 + ((lane >> 4) << 3);
  v8f acc;
  if (accum) {
#pragma unroll
    for (int i = 0; i < 8; ++i) acc[i] = C[(size_t)(rb + i) * ldc + col];
  } else {
    float bv = bias ? bias[col] : 0.0f;
#pragma unroll
    for (int i = 0; i < 8; ++i) acc[i] = bv;
  }
  for (int k0 = 0; k0 < K; k0 += 32) {
    v16bf a = load_frag16(Abase + k0, lda, lane);
    v16bf b = load_frag16(Wbase + k0, ldw, lane);
    acc = __builtin_amdgcn_wmma_f32_16x16x32_bf16(false, a, false, b,
                                                  (short)0, acc, false, false);
  }
#pragma unroll
  for (int i = 0; i < 8; ++i) C[(size_t)(rb + i) * ldc + col] = acc[i];
}

// ---------------------------------------------------------------------------
// Utility kernels
// ---------------------------------------------------------------------------
__global__ void zero_kernel(float* p, int n) {
  int i = blockIdx.x * blockDim.x + threadIdx.x;
  if (i < n) p[i] = 0.0f;
}
__global__ void convert_kernel(const float* __restrict__ s, bf16_t* __restrict__ d, int n) {
  int i = blockIdx.x * blockDim.x + threadIdx.x;
  if (i < n) d[i] = (bf16_t)s[i];
}
// xt = x[:, t, :].reshape(F, B).T   (torch layout quirk), all steps, to bf16.
__global__ void permute_x_kernel(const float* __restrict__ x, bf16_t* __restrict__ xt_all) {
  int idx = blockIdx.x * blockDim.x + threadIdx.x;     // [0, S*B*F)
  if (idx >= S_ * B_ * F_) return;
  int f = idx & (F_ - 1);
  int b = (idx >> 9) & (B_ - 1);
  int t = idx >> 15;
  int flat = f * B_ + b;
  int bp = flat / F_, fp = flat % F_;
  xt_all[idx] = (bf16_t)x[((size_t)bp * S_ + t) * F_ + fp];
}
__global__ void addrsq_kernel(const float* __restrict__ addr, float* __restrict__ addrsq) {
  int n = blockIdx.x * blockDim.x + threadIdx.x;
  if (n < N_) {
    float s = 0.f;
    for (int a = 0; a < A_; ++a) { float v = addr[n * A_ + a]; s += v * v; }
    addrsq[n] = s;
  }
}

__device__ __forceinline__ float wave_sum32(float v) {
#pragma unroll
  for (int off = 16; off; off >>= 1) v += __shfl_xor(v, off, 32);
  return v;
}

// beta = softplus(h.u_sh + b_sh)+1 ; gamma = sigmoid(h.u_lru + b_lru); qnorm = ||q||
__global__ void scalars_kernel(const float* __restrict__ h, const float* __restrict__ q,
                               const float* __restrict__ u_sh, const float* __restrict__ b_sh,
                               const float* __restrict__ u_lr, const float* __restrict__ b_lr,
                               float* beta, float* gamma, float* qnorm) {
  int b = blockIdx.x, tid = threadIdx.x, wave = tid >> 5, lane = tid & 31;
  __shared__ float r1[8], r2[8], r3[8];
  float d1 = 0.f, d2 = 0.f;
  for (int j = tid; j < H_; j += 256) {
    float hv = h[(size_t)b * H_ + j];
    d1 += hv * u_sh[j];
    d2 += hv * u_lr[j];
  }
  float qv = q[(size_t)b * M_ + tid];
  float d3 = qv * qv;
  d1 = wave_sum32(d1); d2 = wave_sum32(d2); d3 = wave_sum32(d3);
  if (lane == 0) { r1[wave] = d1; r2[wave] = d2; r3[wave] = d3; }
  __syncthreads();
  if (tid == 0) {
    float s1 = 0.f, s2 = 0.f, s3 = 0.f;
    for (int i = 0; i < 8; ++i) { s1 += r1[i]; s2 += r2[i]; s3 += r3[i]; }
    s1 += b_sh[0]; s2 += b_lr[0];
    float sp = (s1 > 20.f) ? s1 : log1pf(expf(s1));
    beta[b]  = sp + 1.0f;
    gamma[b] = 1.0f / (1.0f + expf(-s2));
    qnorm[b] = sqrtf(s3);
  }
}

// Softmax over 512 values, one value per thread (block = 512 = 16 waves).
__device__ __forceinline__ float block_softmax_512(float s, int tid, float* red) {
  int wave = tid >> 5, lane = tid & 31;
  float v = s;
#pragma unroll
  for (int off = 16; off; off >>= 1) v = fmaxf(v, __shfl_xor(v, off, 32));
  if (lane == 0) red[wave] = v;
  __syncthreads();
  if (tid < 16) {
    float x = red[tid];
#pragma unroll
    for (int off = 8; off; off >>= 1) x = fmaxf(x, __shfl_xor(x, off, 32));
    if (tid == 0) red[16] = x;
  }
  __syncthreads();
  float mx = red[16];
  __syncthreads();
  float ex = expf(s - mx);
  float sv = wave_sum32(ex);
  if (lane == 0) red[wave] = sv;
  __syncthreads();
  if (tid < 16) {
    float x = red[tid];
#pragma unroll
    for (int off = 8; off; off >>= 1) x += __shfl_xor(x, off, 32);
    if (tid == 0) red[16] = x;
  }
  __syncthreads();
  float total = red[16];
  __syncthreads();
  return ex / total;
}

// One pass over mem[b]: dots + ||mem row||^2, then sim, double EMA update,
// and both softmaxes (w_read, w_upd). Block = 512 threads, grid = B.
__global__ void addr_sim_kernel(const float* __restrict__ mem, const float* __restrict__ q,
                                const float* __restrict__ addr, const float* __restrict__ addrsq,
                                const float* __restrict__ beta, const float* __restrict__ gamma,
                                const float* __restrict__ qnorm, float* __restrict__ ema,
                                float* __restrict__ w_read, float* __restrict__ w_upd) {
  int b = blockIdx.x, tid = threadIdx.x, wave = tid >> 5, lane = tid & 31;
  __shared__ float qs[M_];
  __shared__ float dots_s[N_];
  __shared__ float sq_s[N_];
  __shared__ float red[17];
  if (tid < M_) qs[tid] = q[(size_t)b * M_ + tid];
  __syncthreads();
  const float* memb = mem + (size_t)b * N_ * C_;
  for (int it = 0; it < 32; ++it) {
    int n = it * 16 + wave;
    const float* row = memb + (size_t)n * C_;
    float d = 0.f, sq = 0.f;
#pragma unroll
    for (int i = 0; i < 7; ++i) {               // 7*32 = 224 = C_
      float v = row[lane + 32 * i];
      d  += v * qs[A_ + lane + 32 * i];
      sq += v * v;
    }
    d += addr[n * A_ + lane] * qs[lane];        // address part (A_ = 32 = lanes)
#pragma unroll
    for (int off = 16; off; off >>= 1) {
      d  += __shfl_xor(d, off, 32);
      sq += __shfl_xor(sq, off, 32);
    }
    if (lane == 0) { dots_s[n] = d; sq_s[n] = sq; }
  }
  __syncthreads();
  int n = tid;
  float be = beta[b], ga = gamma[b], qn = qnorm[b];
  float nf = sqrtf(addrsq[n] + sq_s[n]);
  float denom = fmaxf(nf, EPS_) * fmaxf(qn, EPS_);
  float sim = be * dots_s[n] / denom;
  float e0 = ema[(size_t)b * N_ + n];
  float s1 = sim - ga * e0;
  float e1 = 0.1f * e0 + 0.9f * sim;
  float s2 = sim - ga * e1;
  float e2 = 0.1f * e1 + 0.9f * sim;
  ema[(size_t)b * N_ + n] = e2;
  float wr = block_softmax_512(s1, tid, red);
  float wu = block_softmax_512(s2, tid, red);
  w_read[(size_t)b * N_ + n] = wr;
  w_upd [(size_t)b * N_ + n] = wu;
}

// Second pass over mem[b]: reading (content part), mem <- (1-wu)*mem + wu*relu(cand),
// plus address part of reading. Block = 224 threads, grid = B.
__global__ void read_update_kernel(float* __restrict__ mem, const float* __restrict__ w_read,
                                   const float* __restrict__ w_upd, const float* __restrict__ cand,
                                   const float* __restrict__ addr, float* __restrict__ reading,
                                   bf16_t* __restrict__ reading_bf) {
  int b = blockIdx.x, tid = threadIdx.x;
  __shared__ float wr[N_], wu[N_];
  for (int i = tid; i < N_; i += 224) {
    wr[i] = w_read[(size_t)b * N_ + i];
    wu[i] = w_upd [(size_t)b * N_ + i];
  }
  __syncthreads();
  float cv = fmaxf(cand[(size_t)b * C_ + tid], 0.0f);   // relu
  float racc = 0.f;
  float* mb = mem + (size_t)b * N_ * C_ + tid;
  for (int n = 0; n < N_; ++n) {
    float v = mb[(size_t)n * C_];
    racc += wr[n] * v;
    mb[(size_t)n * C_] = (1.0f - wu[n]) * v + wu[n] * cv;
  }
  reading   [(size_t)b * M_ + A_ + tid] = racc;
  reading_bf[(size_t)b * M_ + A_ + tid] = (bf16_t)racc;
  if (tid < A_) {
    float ra = 0.f;
    for (int n = 0; n < N_; ++n) ra += wr[n] * addr[n * A_ + tid];
    reading   [(size_t)b * M_ + tid] = ra;
    reading_bf[(size_t)b * M_ + tid] = (bf16_t)ra;
  }
}

// GRU pointwise: r/z sigmoid, n tanh with r-gated hidden path, h update.
__global__ void gru_kernel(const float* __restrict__ Gr, const float* __restrict__ Gz,
                           const float* __restrict__ Gn1, const float* __restrict__ Gn2,
                           const float* __restrict__ br, const float* __restrict__ bz,
                           const float* __restrict__ bn,
                           float* __restrict__ h, bf16_t* __restrict__ h_bf) {
  int i = blockIdx.x * blockDim.x + threadIdx.x;      // [0, B*H)
  if (i >= B_ * H_) return;
  int j = i & (H_ - 1);
  float r = 1.0f / (1.0f + expf(-(Gr[i] + br[j])));
  float z = 1.0f / (1.0f + expf(-(Gz[i] + bz[j])));
  float n = tanhf(Gn1[i] + r * Gn2[i] + bn[j]);
  float hn = (1.0f - z) * n + z * h[i];
  h[i] = hn;
  h_bf[i] = (bf16_t)hn;
}

// log_softmax over O=256, block = 256 threads, grid = B.
__global__ void logsoftmax_kernel(const float* __restrict__ logits, float* __restrict__ out) {
  int b = blockIdx.x, tid = threadIdx.x, wave = tid >> 5, lane = tid & 31;
  __shared__ float red[9];
  float v = logits[(size_t)b * O_ + tid];
  float m = v;
#pragma unroll
  for (int off = 16; off; off >>= 1) m = fmaxf(m, __shfl_xor(m, off, 32));
  if (lane == 0) red[wave] = m;
  __syncthreads();
  if (tid < 8) {
    float x = red[tid];
#pragma unroll
    for (int off = 4; off; off >>= 1) x = fmaxf(x, __shfl_xor(x, off, 32));
    if (tid == 0) red[8] = x;
  }
  __syncthreads();
  float mx = red[8];
  __syncthreads();
  float ex = expf(v - mx);
  float s = wave_sum32(ex);
  if (lane == 0) red[wave] = s;
  __syncthreads();
  if (tid < 8) {
    float x = red[tid];
#pragma unroll
    for (int off = 4; off; off >>= 1) x += __shfl_xor(x, off, 32);
    if (tid == 0) red[8] = x;
  }
  __syncthreads();
  out[(size_t)b * O_ + tid] = v - mx - logf(red[8]);
}

// ---------------------------------------------------------------------------
extern "C" void kernel_launch(void* const* d_in, const int* in_sizes, int n_in,
                              void* d_out, int out_size, void* d_ws, size_t ws_size,
                              hipStream_t stream) {
  (void)in_sizes; (void)n_in; (void)out_size; (void)ws_size;
  const float* x      = (const float*)d_in[0];
  const float* addr   = (const float*)d_in[1];
  const float* Wq_f   = (const float*)d_in[2];
  const float* bq     = (const float*)d_in[3];
  const float* u_sh   = (const float*)d_in[4];
  const float* b_sh   = (const float*)d_in[5];
  const float* u_lr   = (const float*)d_in[6];
  const float* b_lr   = (const float*)d_in[7];
  const float* Wch_f  = (const float*)d_in[8];
  const float* Wci_f  = (const float*)d_in[9];
  const float* Wir_f  = (const float*)d_in[10];
  const float* Whr_f  = (const float*)d_in[11];
  const float* Wmr_f  = (const float*)d_in[12];
  const float* b_r    = (const float*)d_in[13];
  const float* Wiz_f  = (const float*)d_in[14];
  const float* Whz_f  = (const float*)d_in[15];
  const float* Wmz_f  = (const float*)d_in[16];
  const float* b_z    = (const float*)d_in[17];
  const float* Win_f  = (const float*)d_in[18];
  const float* Whn_f  = (const float*)d_in[19];
  const float* Wmn_f  = (const float*)d_in[20];
  const float* b_n    = (const float*)d_in[21];
  const float* Wo_f   = (const float*)d_in[22];
  const float* b_o    = (const float*)d_in[23];

  char* base = (char*)d_ws;
  size_t off = 0;
  auto alloc = [&](size_t bytes) -> void* {
    off = (off + 255) & ~(size_t)255;
    void* p = base + off;
    off += bytes;
    return p;
  };
  float*  h       = (float*) alloc((size_t)B_ * H_ * 4);
  float*  mem     = (float*) alloc((size_t)B_ * N_ * C_ * 4);
  float*  ema     = (float*) alloc((size_t)B_ * N_ * 4);
  float*  q       = (float*) alloc((size_t)B_ * M_ * 4);
  float*  beta    = (float*) alloc(B_ * 4);
  float*  gamma   = (float*) alloc(B_ * 4);
  float*  qnorm   = (float*) alloc(B_ * 4);
  float*  w_read  = (float*) alloc((size_t)B_ * N_ * 4);
  float*  w_upd   = (float*) alloc((size_t)B_ * N_ * 4);
  float*  reading = (float*) alloc((size_t)B_ * M_ * 4);
  float*  Gr      = (float*) alloc((size_t)B_ * H_ * 4);
  float*  Gz      = (float*) alloc((size_t)B_ * H_ * 4);
  float*  Gn1     = (float*) alloc((size_t)B_ * H_ * 4);
  float*  Gn2     = (float*) alloc((size_t)B_ * H_ * 4);
  float*  Gc      = (float*) alloc((size_t)B_ * C_ * 4);
  float*  logits  = (float*) alloc((size_t)B_ * O_ * 4);
  float*  addrsq  = (float*) alloc(N_ * 4);
  bf16_t* h_bf    = (bf16_t*)alloc((size_t)B_ * H_ * 2);
  bf16_t* rd_bf   = (bf16_t*)alloc((size_t)B_ * M_ * 2);
  bf16_t* xt_all  = (bf16_t*)alloc((size_t)S_ * B_ * F_ * 2);
  bf16_t* wq      = (bf16_t*)alloc((size_t)M_ * H_ * 2);
  bf16_t* wch     = (bf16_t*)alloc((size_t)C_ * H_ * 2);
  bf16_t* wci     = (bf16_t*)alloc((size_t)C_ * F_ * 2);
  bf16_t* wir     = (bf16_t*)alloc((size_t)H_ * F_ * 2);
  bf16_t* whr     = (bf16_t*)alloc((size_t)H_ * H_ * 2);
  bf16_t* wmr     = (bf16_t*)alloc((size_t)H_ * M_ * 2);
  bf16_t* wiz     = (bf16_t*)alloc((size_t)H_ * F_ * 2);
  bf16_t* whz     = (bf16_t*)alloc((size_t)H_ * H_ * 2);
  bf16_t* wmz     = (bf16_t*)alloc((size_t)H_ * M_ * 2);
  bf16_t* win     = (bf16_t*)alloc((size_t)H_ * F_ * 2);
  bf16_t* whn     = (bf16_t*)alloc((size_t)H_ * H_ * 2);
  bf16_t* wmn     = (bf16_t*)alloc((size_t)H_ * M_ * 2);
  bf16_t* wout    = (bf16_t*)alloc((size_t)O_ * C_ * 2);

  auto zero = [&](float* p, size_t n) {
    zero_kernel<<<(unsigned)((n + 255) / 256), 256, 0, stream>>>(p, (int)n);
  };
  auto conv = [&](const float* s, bf16_t* d, size_t n) {
    convert_kernel<<<(unsigned)((n + 255) / 256), 256, 0, stream>>>(s, d, (int)n);
  };
  auto gemm = [&](const bf16_t* A, int lda, const bf16_t* W, int ldw, float* Cc, int ldc,
                  const float* bias, int accum, int mt, int nt, int K) {
    int tiles = mt * nt;
    wmma_gemm_bf16<<<(tiles + 3) / 4, 128, 0, stream>>>(A, lda, W, ldw, Cc, ldc,
                                                        bias, accum, mt, nt, K);
  };

  // ---- prologue: zero state, convert weights, permute inputs -------------
  zero(h, (size_t)B_ * H_);
  zero((float*)h_bf, (size_t)B_ * H_ / 2);
  zero(ema, (size_t)B_ * N_);
  zero(mem, (size_t)B_ * N_ * C_);
  conv(Wq_f,  wq,  (size_t)M_ * H_);
  conv(Wch_f, wch, (size_t)C_ * H_);
  conv(Wci_f, wci, (size_t)C_ * F_);
  conv(Wir_f, wir, (size_t)H_ * F_);
  conv(Whr_f, whr, (size_t)H_ * H_);
  conv(Wmr_f, wmr, (size_t)H_ * M_);
  conv(Wiz_f, wiz, (size_t)H_ * F_);
  conv(Whz_f, whz, (size_t)H_ * H_);
  conv(Wmz_f, wmz, (size_t)H_ * M_);
  conv(Win_f, win, (size_t)H_ * F_);
  conv(Whn_f, whn, (size_t)H_ * H_);
  conv(Wmn_f, wmn, (size_t)H_ * M_);
  conv(Wo_f,  wout,(size_t)O_ * C_);
  permute_x_kernel<<<(S_ * B_ * F_) / 256, 256, 0, stream>>>(x, xt_all);
  addrsq_kernel<<<2, 256, 0, stream>>>(addr, addrsq);

  // ---- scan over time ----------------------------------------------------
  for (int t = 0; t < S_; ++t) {
    const bf16_t* xt = xt_all + (size_t)t * B_ * F_;
    // q = h @ Wq^T + bq                             (64x256, K=1024)
    gemm(h_bf, H_, wq, H_, q, M_, bq, 0, 4, 16, H_);
    // beta/gamma/qnorm
    scalars_kernel<<<B_, 256, 0, stream>>>(h, q, u_sh, b_sh, u_lr, b_lr, beta, gamma, qnorm);
    // sim/EMA/softmaxes (single pass over mem)
    addr_sim_kernel<<<B_, 512, 0, stream>>>(mem, q, addr, addrsq, beta, gamma, qnorm,
                                            ema, w_read, w_upd);
    // cand (pre-relu) = h @ Wch^T + xt @ Wci^T      (64x224)
    gemm(h_bf, H_, wch, H_, Gc, C_, nullptr, 0, 4, 14, H_);
    gemm(xt,   F_, wci, F_, Gc, C_, nullptr, 1, 4, 14, F_);
    // reading + mem update (second pass over mem)
    read_update_kernel<<<B_, 224, 0, stream>>>(mem, w_read, w_upd, Gc, addr, reading, rd_bf);
    // GRU gate GEMMs (64x1024 each)
    gemm(xt,    F_, wir, F_, Gr,  H_, nullptr, 0, 4, 64, F_);
    gemm(h_bf,  H_, whr, H_, Gr,  H_, nullptr, 1, 4, 64, H_);
    gemm(rd_bf, M_, wmr, M_, Gr,  H_, nullptr, 1, 4, 64, M_);
    gemm(xt,    F_, wiz, F_, Gz,  H_, nullptr, 0, 4, 64, F_);
    gemm(h_bf,  H_, whz, H_, Gz,  H_, nullptr, 1, 4, 64, H_);
    gemm(rd_bf, M_, wmz, M_, Gz,  H_, nullptr, 1, 4, 64, M_);
    gemm(xt,    F_, win, F_, Gn1, H_, nullptr, 0, 4, 64, F_);
    gemm(rd_bf, M_, wmn, M_, Gn1, H_, nullptr, 1, 4, 64, M_);
    gemm(h_bf,  H_, whn, H_, Gn2, H_, nullptr, 0, 4, 64, H_);
    // GRU pointwise + h update
    gru_kernel<<<(B_ * H_) / 256, 256, 0, stream>>>(Gr, Gz, Gn1, Gn2, b_r, b_z, b_n, h, h_bf);
  }

  // ---- final output: log_softmax(reading[:, A:] @ Wo^T + b_o) ------------
  gemm(rd_bf + A_, M_, wout, C_, logits, O_, b_o, 0, 4, 16, C_);
  float* out_h   = (float*)d_out;                 // (B,H) first
  float* out_out = out_h + (size_t)B_ * H_;       // then (B,O)
  logsoftmax_kernel<<<B_, 256, 0, stream>>>(logits, out_out);
  hipMemcpyAsync(out_h, h, (size_t)B_ * H_ * 4, hipMemcpyDeviceToDevice, stream);
}
